// MultiRNN_61581241090633
// MI455X (gfx1250) — matrix operand order repaired and checked by reading, not compiled, not run on hardware
//
#include <hip/hip_runtime.h>
#include <hip/hip_bf16.h>
#include <math.h>

#define HH 512
#define NN 1024
#define GG 16
#define VV 32000

typedef __attribute__((ext_vector_type(16))) __bf16 v16bf;
typedef __attribute__((ext_vector_type(8)))  float  v8f;
typedef __attribute__((ext_vector_type(2)))  float  v2f;

union Frag16 { v16bf v; uint4 q[2]; };

// ---------------------------------------------------------------------------
// Kernel 1: group compaction. counts/offsets/permutation for z (N=1024, G=16).
// Deterministic result set per group (intra-group order irrelevant downstream).
// ---------------------------------------------------------------------------
__global__ __launch_bounds__(1024) void k_meta(const int* __restrict__ z,
                                               int* __restrict__ perm,
                                               int* __restrict__ offs) {
  __shared__ int cnt[GG], cur[GG], offs_s[GG + 1];
  int t = threadIdx.x;
  if (t < GG) { cnt[t] = 0; cur[t] = 0; }
  __syncthreads();
  int g = z[t];
  atomicAdd(&cnt[g], 1);
  __syncthreads();
  if (t == 0) {
    int s = 0;
    for (int i = 0; i < GG; ++i) { offs_s[i] = s; s += cnt[i]; }
    offs_s[GG] = s;
  }
  __syncthreads();
  int pos = offs_s[g] + atomicAdd(&cur[g], 1);
  perm[pos] = t;
  if (t <= GG) offs[t] = offs_s[t];
}

// ---------------------------------------------------------------------------
// Kernel 2: a[g][k] = dot(W[g][k,:], E[x0]) for all 16 groups (8.4 MFLOP).
// grid (16 groups, 8 k-blocks of 64), 256 threads: 4 threads per k, each sums
// a 128-wide h-slice; deterministic LDS tree reduce (no f32 atomics).
// ---------------------------------------------------------------------------
__global__ __launch_bounds__(256) void k_we(const int* __restrict__ x,
                                            const float* __restrict__ E,
                                            const float* __restrict__ W,
                                            float* __restrict__ a) {
  __shared__ float e[HH];
  __shared__ float part[256];
  int t = threadIdx.x;
  int g = blockIdx.x;
  int x0 = x[0];
  e[t]       = E[(size_t)x0 * HH + t];
  e[t + 256] = E[(size_t)x0 * HH + t + 256];
  __syncthreads();
  int kl = t >> 2, q = t & 3;
  int k = blockIdx.y * 64 + kl;
  const float* wr = W + ((size_t)g * HH + k) * HH + q * 128;
  const float* ep = e + q * 128;
  float s = 0.f;
#pragma unroll 4
  for (int hh = 0; hh < 128; ++hh) s += wr[hh] * ep[hh];
  part[t] = s;
  __syncthreads();
  if (q == 0)
    a[g * HH + k] = (part[t] + part[t + 1]) + (part[t + 2] + part[t + 3]);
}

// ---------------------------------------------------------------------------
// Kernel 3: split Vw (f32) into hi/lo bf16 pair once (streaming, ~130MB).
// hi = RNE(f) as bf16; lo = RNE(f - hi). hi+lo carries ~16-17 mantissa bits.
// ---------------------------------------------------------------------------
__global__ __launch_bounds__(256) void k_split(const float* __restrict__ src,
                                               __bf16* __restrict__ hi,
                                               __bf16* __restrict__ lo,
                                               int n4) {
  union P { __bf16 b[4]; uint2 u; };
  for (int i = blockIdx.x * blockDim.x + threadIdx.x; i < n4;
       i += gridDim.x * blockDim.x) {
    float4 v = ((const float4*)src)[i];
    float f[4] = {v.x, v.y, v.z, v.w};
    P ph, pl;
#pragma unroll
    for (int j = 0; j < 4; ++j) {
      __bf16 bh = (__bf16)f[j];
      float rem = f[j] - (float)bh;
      ph.b[j] = bh;
      pl.b[j] = (__bf16)rem;
    }
    ((uint2*)hi)[i] = ph.u;
    ((uint2*)lo)[i] = pl.u;
  }
}

// ---------------------------------------------------------------------------
// Kernel 4: h_t = tanh(a[g] + U[g] . h[n]) as grouped GEMM with f32 WMMA
// (16x16x4). One wave per (group, 16-col k-tile, 16-row m-tile). Rows of each
// group gathered via perm. Full f32 precision into tanh. Also emits hi/lo
// bf16 split of h_t for the big GEMM, and f32 h_t straight into d_out.
// ---------------------------------------------------------------------------
__global__ __launch_bounds__(32) void k_ht(const float* __restrict__ h,
                                           const float* __restrict__ U,
                                           const float* __restrict__ a,
                                           const int* __restrict__ perm,
                                           const int* __restrict__ offs,
                                           float* __restrict__ ht,
                                           __bf16* __restrict__ hthi,
                                           __bf16* __restrict__ htlo) {
  int g  = blockIdx.x;       // 16 groups
  int k0 = blockIdx.y * 16;  // 32 output-col tiles
  int mt = blockIdx.z;       // up to 8 row tiles (128 rows/group capacity)
  int off = offs[g], cnt = offs[g + 1] - off;
  if (mt * 16 >= cnt) return;              // uniform per block
  int lane = threadIdx.x;
  int l16 = lane & 15, half = lane >> 4;
  int mrow = mt * 16 + l16;
  int nA = perm[off + (mrow < cnt ? mrow : 0)];
  // A 16x4 f32 layout: lane<16 holds K={0,1}, lane>=16 K={2,3} of row M=lane%16
  const float* hrow = h + (size_t)nA * HH + 2 * half;
  // B 4x16 f32 layout: lane holds column N=lane%16, K pair by lane/16
  const float* urow = U + ((size_t)g * HH + (k0 + l16)) * HH + 2 * half;
  v8f acc = {};
  for (int h0 = 0; h0 < HH; h0 += 4) {
    v2f av = *(const v2f*)(hrow + h0);
    v2f bv = *(const v2f*)(urow + h0);
    acc = __builtin_amdgcn_wmma_f32_16x16x4_f32(false, av, false, bv,
                                                (short)0, acc, false, false);
  }
#pragma unroll
  for (int r = 0; r < 8; ++r) {
    int M = r + 8 * half;  // C/D layout: lanes>=16 hold M=r+8
    int m = mt * 16 + M;
    if (m < cnt) {
      int n = perm[off + m];
      int col = k0 + l16;
      float val = tanhf(acc[r] + a[g * HH + col]);
      ht[(size_t)n * HH + col] = val;
      __bf16 bh = (__bf16)val;
      float rem = val - (float)bh;
      hthi[(size_t)n * HH + col] = bh;
      htlo[(size_t)n * HH + col] = (__bf16)rem;
    }
  }
}

// ---------------------------------------------------------------------------
// Kernel 5: logits = h_t @ Vw^T + Vb  (1024 x 32000 x 512, 33.6 GFLOP).
// bf16x3 split precision on v_wmma_f32_16x16x32_bf16, f32 accumulate.
// WG = 8 waves tiling 128x128; wave = 32x64 = 2x4 WMMA tiles, 16 K-steps.
// Per-lane fragment loads follow ISA VGPR layouts:
//   A (16-bit 16x32): lane<16 -> K {0..7,16..23}; lane>=16 -> K {8..15,24..31}
//   B (16-bit 32x16): lane holds column N=lane%16, same K split as A (row of Vw)
// ---------------------------------------------------------------------------
__global__ __launch_bounds__(256) void k_logits(const __bf16* __restrict__ Ah,
                                                const __bf16* __restrict__ Al,
                                                const __bf16* __restrict__ Bh,
                                                const __bf16* __restrict__ Bl,
                                                const float* __restrict__ Vb,
                                                float* __restrict__ out) {
  int lane = threadIdx.x & 31;
  int wave = threadIdx.x >> 5;
  int wm = wave & 3, wn = wave >> 2;
  int l16 = lane & 15, half = lane >> 4;
  int n0 = blockIdx.y * 128 + wm * 32;
  int v0 = blockIdx.x * 128 + wn * 64;

  size_t ra[2], rb[4];
#pragma unroll
  for (int m = 0; m < 2; ++m) ra[m] = (size_t)(n0 + m * 16 + l16) * HH;
#pragma unroll
  for (int n = 0; n < 4; ++n) rb[n] = (size_t)(v0 + n * 16 + l16) * HH;

  v8f acc[2][4] = {};

  for (int kk = 0; kk < 16; ++kk) {
    int kA = kk * 32 + half * 8;
    int kB = kk * 32 + half * 8;
    Frag16 ah[2], al[2], bh[4], bl[4];
#pragma unroll
    for (int m = 0; m < 2; ++m) {
      ah[m].q[0] = *(const uint4*)(Ah + ra[m] + kA);
      ah[m].q[1] = *(const uint4*)(Ah + ra[m] + kA + 16);
      al[m].q[0] = *(const uint4*)(Al + ra[m] + kA);
      al[m].q[1] = *(const uint4*)(Al + ra[m] + kA + 16);
    }
#pragma unroll
    for (int n = 0; n < 4; ++n) {
      bh[n].q[0] = *(const uint4*)(Bh + rb[n] + kB);
      bh[n].q[1] = *(const uint4*)(Bh + rb[n] + kB + 16);
      bl[n].q[0] = *(const uint4*)(Bl + rb[n] + kB);
      bl[n].q[1] = *(const uint4*)(Bl + rb[n] + kB + 16);
    }
#pragma unroll
    for (int m = 0; m < 2; ++m)
#pragma unroll
      for (int n = 0; n < 4; ++n) {
        acc[m][n] = __builtin_amdgcn_wmma_f32_16x16x32_bf16(
            false, ah[m].v, false, bh[n].v, (short)0, acc[m][n], false, false);
        acc[m][n] = __builtin_amdgcn_wmma_f32_16x16x32_bf16(
            false, ah[m].v, false, bl[n].v, (short)0, acc[m][n], false, false);
        acc[m][n] = __builtin_amdgcn_wmma_f32_16x16x32_bf16(
            false, al[m].v, false, bh[n].v, (short)0, acc[m][n], false, false);
      }
  }

#pragma unroll
  for (int n = 0; n < 4; ++n) {
    float vb = Vb[v0 + n * 16 + l16];
#pragma unroll
    for (int m = 0; m < 2; ++m)
#pragma unroll
      for (int r = 0; r < 8; ++r) {
        int row = n0 + m * 16 + r + half * 8;
        int col = v0 + n * 16 + l16;
        out[(size_t)row * VV + col] = acc[m][n][r] + vb;
      }
  }
}

// ---------------------------------------------------------------------------
extern "C" void kernel_launch(void* const* d_in, const int* in_sizes, int n_in,
                              void* d_out, int out_size, void* d_ws,
                              size_t ws_size, hipStream_t stream) {
  (void)in_sizes; (void)n_in; (void)out_size; (void)ws_size;
  const int*   x  = (const int*)d_in[0];
  const int*   z  = (const int*)d_in[1];
  const float* h  = (const float*)d_in[2];
  const float* E  = (const float*)d_in[3];
  const float* W  = (const float*)d_in[4];
  const float* U  = (const float*)d_in[5];
  const float* Vw = (const float*)d_in[6];
  const float* Vb = (const float*)d_in[7];

  float* out    = (float*)d_out;                 // logits: 1024*32000
  float* ht_out = out + (size_t)NN * VV;         // h_t:    1024*512

  // workspace layout (~68 MB), all 16B aligned
  __bf16* VwHi = (__bf16*)d_ws;
  __bf16* VwLo = VwHi + (size_t)VV * HH;
  __bf16* HtHi = VwLo + (size_t)VV * HH;
  __bf16* HtLo = HtHi + (size_t)NN * HH;
  float*  a    = (float*)(HtLo + (size_t)NN * HH);
  int*    perm = (int*)(a + GG * HH);
  int*    offs = perm + NN;

  k_meta  <<<1, 1024, 0, stream>>>(z, perm, offs);
  k_we    <<<dim3(GG, 8), 256, 0, stream>>>(x, E, W, a);
  k_split <<<4096, 256, 0, stream>>>(Vw, VwHi, VwLo, (VV * HH) / 4);
  k_ht    <<<dim3(GG, HH / 16, 8), 32, 0, stream>>>(h, U, a, perm, offs,
                                                    ht_out, HtHi, HtLo);
  k_logits<<<dim3(VV / 128, NN / 128), 256, 0, stream>>>(HtHi, HtLo, VwHi,
                                                         VwLo, Vb, out);
}
